// BiDAF_8126078124254
// MI455X (gfx1250) — matrix operand order repaired and checked by reading, not compiled, not run on hardware
//
#include <hip/hip_runtime.h>
#include <hip/hip_bf16.h>
#include <math.h>

typedef _Float16 half_t;
typedef __attribute__((ext_vector_type(16))) _Float16 v16h;
typedef __attribute__((ext_vector_type(8)))  float    v8f;

#define B_  32
#define LC_ 512
#define LQ_ 64
#define D_  256
#define H_  128
#define NC_ (B_*LC_)        // 16384 context tokens
#define NQ_ (B_*LQ_)        // 2048 query tokens
#define NT_ (NC_+NQ_)       // 18432 tokens total

__device__ __forceinline__ float sigmoidf_(float x){ return 1.f/(1.f+expf(-x)); }

// ---- CDNA5 WMMA fragment loaders (wave32; this file's contiguous K order, shared by A and B) ----
// A-matrix 16x32 f16: lane(0-15)=M, lane>=16 same M, K contiguous:
//   slot j -> K = (lane>>4)*16 + j
__device__ __forceinline__ v16h frag_a_lds(const half_t* __restrict__ s, int ld){
  int lane = threadIdx.x & 31;
  int m  = lane & 15;
  int kh = (lane >> 4) * 16;
  v16h f;
#pragma unroll
  for (int j = 0; j < 16; ++j){
    int k = j + kh;
    f[j] = s[m*ld + k];
  }
  return f;
}
// B-matrix 32x16 f16: lane&15 = N, slot j -> K = (lane>>4)*16 + j.
// BT=0: B stored (N,K) row-major (weight.T convention). BT=1: (K,N) row-major.
template<int BT>
__device__ __forceinline__ v16h frag_b(const half_t* __restrict__ Bp, int n0, int k0, int ldb){
  int lane = threadIdx.x & 31;
  int n  = n0 + (lane & 15);
  int kb = k0 + ((lane >> 4) << 4);
  v16h f;
#pragma unroll
  for (int j = 0; j < 16; ++j){
    int k = kb + j;
    f[j] = BT ? Bp[(long)k*ldb + n] : Bp[(long)n*ldb + k];
  }
  return f;
}

// ---- generic batched WMMA GEMM: C[M,N] = act(A[M,K]*B + bias) ----
// block = 128 threads (4 waves): tile 16(M) x 64(N), waves across N.
// ACT: 0 none, 1 relu, 2 sigmoid
template<int BT, int ACT>
__global__ __launch_bounds__(128) void
gemm_wmma(const float* __restrict__ A, const half_t* __restrict__ Bw,
          const float* __restrict__ bias, float* __restrict__ C,
          int M, int N, int K, long sA, long sB, long sC)
{
  __shared__ half_t As[16*32];
  int bz = blockIdx.z;
  A  += (long)bz * sA;
  Bw += (long)bz * sB;
  C  += (long)bz * sC;
  int m0   = blockIdx.y * 16;
  int wv   = threadIdx.x >> 5;
  int n0   = blockIdx.x * 64 + wv * 16;
  int lane = threadIdx.x & 31;
  int ldb  = BT ? N : K;
  v8f acc = {0.f,0.f,0.f,0.f,0.f,0.f,0.f,0.f};
  for (int k0 = 0; k0 < K; k0 += 32){
    __syncthreads();
    for (int i = threadIdx.x; i < 512; i += 128){
      int r = i >> 5, c = i & 31;
      As[i] = (half_t)A[(long)(m0 + r)*K + (k0 + c)];
    }
    __syncthreads();
    v16h af = frag_a_lds(As, 32);
    v16h bf = frag_b<BT>(Bw, n0, k0, ldb);
    acc = __builtin_amdgcn_wmma_f32_16x16x32_f16(false, af, false, bf, (short)0, acc, false, false);
  }
  int n  = n0 + (lane & 15);
  int mo = m0 + ((lane >> 4) << 3);
  float bv = bias ? bias[n] : 0.f;
#pragma unroll
  for (int r = 0; r < 8; ++r){
    float v = acc[r] + bv;
    if (ACT == 1) v = v > 0.f ? v : 0.f;
    if (ACT == 2) v = sigmoidf_(v);
    C[(long)(mo + r)*N + n] = v;
  }
}

// ---- persistent BiLSTM recurrence: one block per direction, 16 waves ----
// gates_x precomputed (x @ Wih^T), biases folded in the elementwise stage.
// Whh lives in VGPRs: each wave preloads its 4 tile-columns (4x4 v16h = 128
// VGPRs/thread) before the time loop, so the 512-step recurrence touches only
// LDS (h state, gate staging) + v_wmma. __launch_bounds__(512,1) grants the
// 256-VGPR budget so nothing spills to scratch.
// LDS: h f16 32x128 (8KB) | c f32 32x128 (16KB) | gates f32 32x512 (64KB) = 88KB dynamic
#define LSTM_SMEM (88*1024)
__global__ __launch_bounds__(512, 1) void
lstm_rec(const float* __restrict__ gxF, const float* __restrict__ gxB,
         const half_t* __restrict__ whF, const half_t* __restrict__ whB,
         const float* __restrict__ bihF, const float* __restrict__ bhhF,
         const float* __restrict__ bihB, const float* __restrict__ bhhB,
         float* __restrict__ out, int T)
{
  extern __shared__ char smem[];
  half_t* hb = (half_t*)smem;                 // 32x128 f16
  float*  cb = (float*)(smem + 8*1024);       // 32x128 f32
  float*  gb = (float*)(smem + 24*1024);      // 32x512 f32
  const int dir = blockIdx.x;
  const float*  gx = dir ? gxB : gxF;
  const half_t* wh = dir ? whB : whF;
  const float*  bi = dir ? bihB : bihF;
  const float*  bh = dir ? bhhB : bhhF;
  int tid = threadIdx.x;
  for (int i = tid; i < 32*128; i += 512){ hb[i] = (half_t)0.f; cb[i] = 0.f; }
  int wv = tid >> 5, lane = tid & 31;

  // explicit register residency for this wave's Whh tiles (loop-invariant)
  v16h breg[4][4];
#pragma unroll
  for (int j = 0; j < 4; ++j){
    int nt = (wv*4 + j) & 31;
#pragma unroll
    for (int kk = 0; kk < 4; ++kk)
      breg[j][kk] = frag_b<0>(wh, nt*16, kk*32, 128);
  }
  __syncthreads();

  for (int s = 0; s < T; ++s){
    int t = dir ? (T - 1 - s) : s;
    // gates += h @ Whh^T : M=32(2 tiles) x N=512(32 tiles), K=128(4 wmma steps)
#pragma unroll
    for (int j = 0; j < 4; ++j){
      int tile = wv*4 + j;
      int mt = tile >> 5, nt = tile & 31;
      v8f acc = {0.f,0.f,0.f,0.f,0.f,0.f,0.f,0.f};
#pragma unroll
      for (int kk = 0; kk < 4; ++kk){
        v16h af = frag_a_lds(hb + mt*16*128 + kk*32, 128);
        acc = __builtin_amdgcn_wmma_f32_16x16x32_f16(false, af, false, breg[j][kk],
                                                     (short)0, acc, false, false);
      }
      int n  = nt*16 + (lane & 15);
      int mo = mt*16 + ((lane >> 4) << 3);
#pragma unroll
      for (int r = 0; r < 8; ++r) gb[(mo + r)*512 + n] = acc[r];
    }
    __syncthreads();
    // elementwise LSTM cell update (i,f,g,o split along N)
    for (int i = tid; i < 32*128; i += 512){
      int bb = i >> 7, hI = i & 127;
      long base = ((long)bb*T + t) * 512;
      float gi = gb[bb*512 +       hI] + gx[base +       hI] + bi[      hI] + bh[      hI];
      float gf = gb[bb*512 + 128 + hI] + gx[base + 128 + hI] + bi[128 + hI] + bh[128 + hI];
      float gg = gb[bb*512 + 256 + hI] + gx[base + 256 + hI] + bi[256 + hI] + bh[256 + hI];
      float go = gb[bb*512 + 384 + hI] + gx[base + 384 + hI] + bi[384 + hI] + bh[384 + hI];
      float cc = sigmoidf_(gf)*cb[i] + sigmoidf_(gi)*tanhf(gg);
      float hh = sigmoidf_(go)*tanhf(cc);
      cb[i] = cc;
      hb[i] = (half_t)hh;
      out[((long)bb*T + t)*256 + dir*128 + hI] = hh;
    }
    __syncthreads();
  }
}

// ---- small elementwise / reduction kernels ----
__global__ void gather_embed(const int* __restrict__ cw, const int* __restrict__ qw,
                             const float* __restrict__ emb, float* __restrict__ X)
{
  long idx = (long)blockIdx.x*blockDim.x + threadIdx.x;
  if (idx >= (long)NT_*D_) return;
  int row = (int)(idx >> 8), d = (int)(idx & 255);
  int word = (row < NC_) ? cw[row] : qw[row - NC_];
  X[idx] = emb[(long)word*D_ + d];
}

__global__ void hw_combine(float* __restrict__ X, const float* __restrict__ Hh,
                           const float* __restrict__ G, long n)
{
  long i = (long)blockIdx.x*blockDim.x + threadIdx.x;
  if (i >= n) return;
  float g = G[i];
  X[i] = g*Hh[i] + (1.f - g)*X[i];
}

__global__ void f32_to_f16(const float* __restrict__ s, half_t* __restrict__ d, long n)
{
  long i = (long)blockIdx.x*blockDim.x + threadIdx.x;
  if (i < n) d[i] = (half_t)s[i];
}

__global__ void rowdot(const float* __restrict__ X, const float* __restrict__ w,
                       float* __restrict__ out, int rows, int K)
{
  int r = blockIdx.x*blockDim.x + threadIdx.x;
  if (r >= rows) return;
  const float* x = X + (long)r*K;
  float acc = 0.f;
  for (int k = 0; k < K; ++k) acc += x[k]*w[k];
  out[r] = acc;
}

__global__ void cprime_k(const float* __restrict__ cL, const float* __restrict__ wcq,
                         float* __restrict__ cp, long n)
{
  long i = (long)blockIdx.x*blockDim.x + threadIdx.x;
  if (i < n) cp[i] = cL[i]*wcq[i & 255];
}

// per (b,i) row of 64: add cw/qw/scalar biases, softmax in place, record pre-softmax row max
__global__ void attn_softmax(float* __restrict__ s, const float* __restrict__ cw,
                             const float* __restrict__ qw, const float* __restrict__ bc,
                             const float* __restrict__ bq, const float* __restrict__ bcq,
                             float* __restrict__ rowmax)
{
  __shared__ float red[64];
  int bid = blockIdx.x;            // b*512 + i
  int b   = bid >> 9;
  int j   = threadIdx.x;           // 0..63
  float v = s[(long)bid*64 + j] + cw[bid] + qw[b*64 + j] + (*bc + *bq + *bcq);
  red[j] = v; __syncthreads();
  for (int o = 32; o > 0; o >>= 1){ if (j < o) red[j] = fmaxf(red[j], red[j+o]); __syncthreads(); }
  float mx = red[0]; __syncthreads();
  float e = expf(v - mx);
  red[j] = e; __syncthreads();
  for (int o = 32; o > 0; o >>= 1){ if (j < o) red[j] += red[j+o]; __syncthreads(); }
  float sum = red[0];
  s[(long)bid*64 + j] = e/sum;
  if (j == 0) rowmax[bid] = mx;
}

__global__ void softmax_c(const float* __restrict__ rowmax, float* __restrict__ batt)
{
  __shared__ float red[256];
  int b = blockIdx.x, tid = threadIdx.x;
  float m = -1e30f;
  for (int i = tid; i < 512; i += 256) m = fmaxf(m, rowmax[b*512 + i]);
  red[tid] = m; __syncthreads();
  for (int o = 128; o > 0; o >>= 1){ if (tid < o) red[tid] = fmaxf(red[tid], red[tid+o]); __syncthreads(); }
  float mx = red[0]; __syncthreads();
  float sm = 0.f;
  for (int i = tid; i < 512; i += 256) sm += expf(rowmax[b*512 + i] - mx);
  red[tid] = sm; __syncthreads();
  for (int o = 128; o > 0; o >>= 1){ if (tid < o) red[tid] += red[tid+o]; __syncthreads(); }
  float sum = red[0];
  for (int i = tid; i < 512; i += 256)
    batt[b*512 + i] = expf(rowmax[b*512 + i] - mx)/sum;
}

__global__ void q2c_k(const float* __restrict__ batt, const float* __restrict__ cL,
                      float* __restrict__ q2c)
{
  int b = blockIdx.x, d = threadIdx.x;   // 256 threads
  float acc = 0.f;
  for (int i = 0; i < 512; ++i) acc += batt[b*512 + i]*cL[((long)b*512 + i)*256 + d];
  q2c[b*256 + d] = acc;
}

__global__ void build_g(const float* __restrict__ cL, const float* __restrict__ c2q,
                        const float* __restrict__ q2c, float* __restrict__ G)
{
  long idx = (long)blockIdx.x*blockDim.x + threadIdx.x;
  if (idx >= (long)NC_*256) return;
  int row = (int)(idx >> 8), d = (int)(idx & 255), b = row >> 9;
  float c = cL[idx], a = c2q[idx];
  long g0 = (long)row*1024 + d;
  G[g0      ] = c;
  G[g0 + 256] = a;
  G[g0 + 512] = c*a;
  G[g0 + 768] = c*q2c[b*256 + d];
}

__global__ void pout(const float* __restrict__ G, const float* __restrict__ M,
                     const float* __restrict__ w, const float* __restrict__ bptr,
                     float* __restrict__ out)
{
  int r = blockIdx.x*blockDim.x + threadIdx.x;
  if (r >= NC_) return;
  float acc = *bptr;
  const float* g = G + (long)r*1024;
  for (int k = 0; k < 1024; ++k) acc += g[k]*w[k];
  const float* m = M + (long)r*256;
  for (int k = 0; k < 256; ++k) acc += m[k]*w[1024 + k];
  out[r] = acc;
}

// ---- host side ----
static inline void cvt16(const float* s, half_t* d, long n, hipStream_t st){
  f32_to_f16<<<(unsigned)((n + 255)/256), 256, 0, st>>>(s, d, n);
}
template<int BT, int ACT>
static inline void gemm(const float* A, const half_t* Bw, const float* bias, float* C,
                        int M, int N, int K, int batch, long sA, long sB, long sC,
                        hipStream_t st){
  dim3 g(N/64, M/16, batch);
  gemm_wmma<BT,ACT><<<g, 128, 0, st>>>(A, Bw, bias, C, M, N, K, sA, sB, sC);
}

extern "C" void kernel_launch(void* const* d_in, const int* in_sizes, int n_in,
                              void* d_out, int out_size, void* d_ws, size_t ws_size,
                              hipStream_t stream)
{
  (void)in_sizes; (void)n_in; (void)out_size; (void)ws_size;
  // input order (setup_inputs flattened)
  const int*   c_word = (const int*)d_in[0];
  const int*   q_word = (const int*)d_in[1];
  const float* emb    = (const float*)d_in[4];
  const float* hw_wl[2] = { (const float*)d_in[5],  (const float*)d_in[9]  };
  const float* hw_bl[2] = { (const float*)d_in[6],  (const float*)d_in[10] };
  const float* hw_wg[2] = { (const float*)d_in[7],  (const float*)d_in[11] };
  const float* hw_bg[2] = { (const float*)d_in[8],  (const float*)d_in[12] };
  const float* att_wc  = (const float*)d_in[13];
  const float* att_bc  = (const float*)d_in[14];
  const float* att_wq  = (const float*)d_in[15];
  const float* att_bq  = (const float*)d_in[16];
  const float* att_wcq = (const float*)d_in[17];
  const float* att_bcq = (const float*)d_in[18];
  // lstm param blocks: ctx@19, mod1@27, mod2@35, out@43 ; each: wihF,whhF,bihF,bhhF,wihB,whhB,bihB,bhhB
  const int lb[4] = {19, 27, 35, 43};
  const int lK[4] = {256, 1024, 256, 256};     // input dims
  const float* p1w = (const float*)d_in[51];
  const float* p1b = (const float*)d_in[52];
  const float* p2w = (const float*)d_in[53];
  const float* p2b = (const float*)d_in[54];

  // workspace carve-out
  uint8_t* base = (uint8_t*)d_ws;
  size_t off = 0;
  auto alloc = [&](size_t bytes)->void*{
    void* p = base + off;
    off += (bytes + 255) & ~(size_t)255;
    return p;
  };
  float*  X    = (float*)alloc((size_t)NT_*256*4);
  float*  Ht   = (float*)alloc((size_t)NT_*256*4);
  float*  Gt   = (float*)alloc((size_t)NT_*256*4);
  float*  gxF  = (float*)alloc((size_t)NC_*512*4);
  float*  gxB  = (float*)alloc((size_t)NC_*512*4);
  float*  gxqF = (float*)alloc((size_t)NQ_*512*4);
  float*  gxqB = (float*)alloc((size_t)NQ_*512*4);
  float*  cL   = (float*)alloc((size_t)NC_*256*4);
  float*  qL   = (float*)alloc((size_t)NQ_*256*4);
  float*  cw   = (float*)alloc((size_t)NC_*4);
  float*  qwv  = (float*)alloc((size_t)NQ_*4);
  float*  cp   = (float*)alloc((size_t)NC_*256*4);
  half_t* qh16 = (half_t*)alloc((size_t)NQ_*256*2);
  float*  sAtt = (float*)alloc((size_t)B_*LC_*LQ_*4);   // s, then a in place
  float*  rmax = (float*)alloc((size_t)NC_*4);
  float*  batt = (float*)alloc((size_t)NC_*4);
  float*  q2c  = (float*)alloc((size_t)B_*256*4);
  float*  c2q  = (float*)alloc((size_t)NC_*256*4);
  float*  G    = (float*)alloc((size_t)NC_*1024*4);
  float*  m1   = (float*)alloc((size_t)NC_*256*4);
  float*  mo   = (float*)alloc((size_t)NC_*256*4);
  float*  m2   = (float*)alloc((size_t)NC_*256*4);
  half_t* hwl16[2], *hwg16[2];
  for (int l = 0; l < 2; ++l){
    hwl16[l] = (half_t*)alloc((size_t)256*256*2);
    hwg16[l] = (half_t*)alloc((size_t)256*256*2);
  }
  half_t* wih16[4][2]; half_t* whh16[4][2];
  for (int l = 0; l < 4; ++l)
    for (int d = 0; d < 2; ++d){
      wih16[l][d] = (half_t*)alloc((size_t)512*lK[l]*2);
      whh16[l][d] = (half_t*)alloc((size_t)512*128*2);
    }

  (void)hipFuncSetAttribute((const void*)lstm_rec,
                            hipFuncAttributeMaxDynamicSharedMemorySize, LSTM_SMEM);

  // 1) f16 weight conversion (one-time per launch; tiny vs compute)
  for (int l = 0; l < 2; ++l){
    cvt16(hw_wl[l], hwl16[l], 256*256, stream);
    cvt16(hw_wg[l], hwg16[l], 256*256, stream);
  }
  for (int l = 0; l < 4; ++l)
    for (int d = 0; d < 2; ++d){
      cvt16((const float*)d_in[lb[l] + 4*d    ], wih16[l][d], (long)512*lK[l], stream);
      cvt16((const float*)d_in[lb[l] + 4*d + 1], whh16[l][d], (long)512*128,   stream);
    }

  // 2) embedding gather (c ‖ q tokens)
  gather_embed<<<(unsigned)(((long)NT_*256 + 255)/256), 256, 0, stream>>>(c_word, q_word, emb, X);

  // 3) highway x2: h=relu(xWl^T+bl), g=sigmoid(xWg^T+bg), x=g*h+(1-g)*x
  for (int l = 0; l < 2; ++l){
    gemm<0,1>(X, hwl16[l], hw_bl[l], Ht, NT_, 256, 256, 1, 0,0,0, stream);
    gemm<0,2>(X, hwg16[l], hw_bg[l], Gt, NT_, 256, 256, 1, 0,0,0, stream);
    hw_combine<<<(unsigned)(((long)NT_*256 + 255)/256), 256, 0, stream>>>(X, Ht, Gt, (long)NT_*256);
  }

  // 4) context BiLSTM: hoist input projections, then persistent recurrence
  gemm<0,0>(X,            wih16[0][0], nullptr, gxF,  NC_, 512, 256, 1, 0,0,0, stream);
  gemm<0,0>(X,            wih16[0][1], nullptr, gxB,  NC_, 512, 256, 1, 0,0,0, stream);
  gemm<0,0>(X + (long)NC_*256, wih16[0][0], nullptr, gxqF, NQ_, 512, 256, 1, 0,0,0, stream);
  gemm<0,0>(X + (long)NC_*256, wih16[0][1], nullptr, gxqB, NQ_, 512, 256, 1, 0,0,0, stream);
  lstm_rec<<<2, 512, LSTM_SMEM, stream>>>(gxF, gxB, whh16[0][0], whh16[0][1],
      (const float*)d_in[21], (const float*)d_in[22], (const float*)d_in[25], (const float*)d_in[26],
      cL, LC_);
  lstm_rec<<<2, 512, LSTM_SMEM, stream>>>(gxqF, gxqB, whh16[0][0], whh16[0][1],
      (const float*)d_in[21], (const float*)d_in[22], (const float*)d_in[25], (const float*)d_in[26],
      qL, LQ_);

  // 5) attention: s = cw[:,None] + qw[None,:] + (c*wcq)@q^T + consts
  rowdot<<<(NC_ + 255)/256, 256, 0, stream>>>(cL, att_wc, cw, NC_, 256);
  rowdot<<<(NQ_ + 255)/256, 256, 0, stream>>>(qL, att_wq, qwv, NQ_, 256);
  cprime_k<<<(unsigned)(((long)NC_*256 + 255)/256), 256, 0, stream>>>(cL, att_wcq, cp, (long)NC_*256);
  cvt16(qL, qh16, (long)NQ_*256, stream);
  gemm<0,0>(cp, qh16, nullptr, sAtt, LC_, LQ_, 256, B_,
            (long)LC_*256, (long)LQ_*256, (long)LC_*LQ_, stream);
  attn_softmax<<<NC_, 64, 0, stream>>>(sAtt, cw, qwv, att_bc, att_bq, att_bcq, rmax);
  // c2q = a @ q  (B stored (K=LQ, N=256) row-major -> BT=1)
  gemm<1,0>(sAtt, qh16, nullptr, c2q, LC_, 256, LQ_, B_,
            (long)LC_*LQ_, (long)LQ_*256, (long)LC_*256, stream);
  softmax_c<<<B_, 256, 0, stream>>>(rmax, batt);
  q2c_k<<<B_, 256, 0, stream>>>(batt, cL, q2c);

  // 6) g = [c, c2q, c*c2q, c*q2c]
  build_g<<<(unsigned)(((long)NC_*256 + 255)/256), 256, 0, stream>>>(cL, c2q, q2c, G);

  // 7) modeling BiLSTM 1 (K=1024)
  gemm<0,0>(G, wih16[1][0], nullptr, gxF, NC_, 512, 1024, 1, 0,0,0, stream);
  gemm<0,0>(G, wih16[1][1], nullptr, gxB, NC_, 512, 1024, 1, 0,0,0, stream);
  lstm_rec<<<2, 512, LSTM_SMEM, stream>>>(gxF, gxB, whh16[1][0], whh16[1][1],
      (const float*)d_in[29], (const float*)d_in[30], (const float*)d_in[33], (const float*)d_in[34],
      m1, LC_);

  // 8) modeling BiLSTM 2 -> m
  gemm<0,0>(m1, wih16[2][0], nullptr, gxF, NC_, 512, 256, 1, 0,0,0, stream);
  gemm<0,0>(m1, wih16[2][1], nullptr, gxB, NC_, 512, 256, 1, 0,0,0, stream);
  lstm_rec<<<2, 512, LSTM_SMEM, stream>>>(gxF, gxB, whh16[2][0], whh16[2][1],
      (const float*)d_in[37], (const float*)d_in[38], (const float*)d_in[41], (const float*)d_in[42],
      mo, LC_);

  // 9) output BiLSTM -> m2
  gemm<0,0>(mo, wih16[3][0], nullptr, gxF, NC_, 512, 256, 1, 0,0,0, stream);
  gemm<0,0>(mo, wih16[3][1], nullptr, gxB, NC_, 512, 256, 1, 0,0,0, stream);
  lstm_rec<<<2, 512, LSTM_SMEM, stream>>>(gxF, gxB, whh16[3][0], whh16[3][1],
      (const float*)d_in[45], (const float*)d_in[46], (const float*)d_in[49], (const float*)d_in[50],
      m2, LC_);

  // 10) p1 = [g, m] . p1_w + b ; p2 = [g, m2] . p2_w + b
  float* out = (float*)d_out;
  pout<<<(NC_ + 255)/256, 256, 0, stream>>>(G, mo, p1w, p1b, out);
  pout<<<(NC_ + 255)/256, 256, 0, stream>>>(G, m2, p2w, p2b, out + NC_);
}
